// GroupQueryAttention_71734543778301
// MI455X (gfx1250) — matrix operand order repaired and checked by reading, not compiled, not run on hardware
//
#include <hip/hip_runtime.h>
#include <hip/hip_bf16.h>

// ---------------------------------------------------------------------------
// GQA attention for MI455X (gfx1250, wave32, WMMA + TDM + async-LDS).
// B=2, S=2048, D=2048, H=16, G=4, DH=128, KV=512.
//   GEMM1: QKV = X @ [Wq|Wk|Wv] + b   (bf16 WMMA, TDM tile staging)
//   Flash attention per (b,h,64-query tile): WMMA QK^T + PV, async K staging
//   GEMM2: out = AO @ Wo + bo          (f32 out)
// ---------------------------------------------------------------------------

typedef __attribute__((ext_vector_type(16))) __bf16   v16bf;
typedef __attribute__((ext_vector_type(8)))  __bf16   v8bf;
typedef __attribute__((ext_vector_type(8)))  float    v8f;
typedef __attribute__((ext_vector_type(4)))  uint32_t u32x4;
typedef __attribute__((ext_vector_type(8)))  uint32_t u32x8;

#define WMMA_BF16(A, Bm, C) \
  __builtin_amdgcn_wmma_f32_16x16x32_bf16(false, (A), false, (Bm), (short)0, (C), false, false)

#if __has_builtin(__builtin_amdgcn_global_load_async_to_lds_b128)
#define HAS_ASYNC_LDS 1
typedef int v4i_b128 __attribute__((vector_size(16)));
typedef __attribute__((address_space(1))) v4i_b128* gptr_b128;
typedef __attribute__((address_space(3))) v4i_b128* lptr_b128;
#endif

__device__ __forceinline__ v8f zero8() {
  v8f z;
#pragma unroll
  for (int i = 0; i < 8; ++i) z[i] = 0.0f;
  return z;
}

__device__ __forceinline__ v16bf join16(v8bf lo, v8bf hi) {
  v16bf a;
#pragma unroll
  for (int i = 0; i < 8; ++i) { a[i] = lo[i]; a[8 + i] = hi[i]; }
  return a;
}

__device__ __forceinline__ void wait_tensorcnt0() {
#if __has_builtin(__builtin_amdgcn_s_wait_tensorcnt)
  __builtin_amdgcn_s_wait_tensorcnt(0);
#else
  asm volatile("s_wait_tensorcnt 0x0" ::: "memory");
#endif
}

// Tensor Data Mover: DMA a 2D tile (tile_w x tile_h elements of 2 bytes) from
// a row-major global tensor (row stride in elements) into LDS at lds_off.
// D# layout per CDNA5 ISA. Issued once per wave.
__device__ __forceinline__ void tdm_load_tile_2d_bf16(
    const void* gaddr, uint32_t lds_off, uint32_t tile_w, uint32_t tile_h,
    uint32_t tensor_w, uint32_t tensor_h, uint32_t row_stride) {
  uint64_t ga = (uint64_t)(uintptr_t)gaddr;
  u32x4 g0;
  g0[0] = 1u;                                   // count=1, user descriptor
  g0[1] = lds_off;                              // lds_addr (bytes)
  g0[2] = (uint32_t)ga;                         // global_addr[31:0]
  g0[3] = (uint32_t)((ga >> 32) & 0x01FFFFFFu)  // global_addr[56:32]
          | 0x80000000u;                        // type=2 ("image")
  u32x8 g1;
  g1[0] = 1u << 16;                             // data_size=1 (2 bytes)
  g1[1] = (tensor_w & 0xFFFFu) << 16;           // tensor_dim0[15:0]
  g1[2] = ((tensor_w >> 16) & 0xFFFFu) | ((tensor_h & 0xFFFFu) << 16);
  g1[3] = ((tensor_h >> 16) & 0xFFFFu) | ((tile_w & 0xFFFFu) << 16);
  g1[4] = tile_h & 0xFFFFu;                     // tile_dim1, tile_dim2=0
  g1[5] = row_stride;                           // tensor_dim0_stride[31:0]
  g1[6] = 0u;
  g1[7] = 0u;
  asm volatile("tensor_load_to_lds %0, %1" :: "s"(g0), "s"(g1) : "memory");
}

// ---------------------------------------------------------------------------
// Conversion / packing kernels
// ---------------------------------------------------------------------------
__global__ void f32_to_bf16_kernel(const float* __restrict__ in,
                                   __bf16* __restrict__ out, int n) {
  int i = blockIdx.x * blockDim.x + threadIdx.x;
  if (i < n) out[i] = (__bf16)in[i];
}

__global__ void build_wqkv_kernel(const float* __restrict__ Wq,
                                  const float* __restrict__ Wk,
                                  const float* __restrict__ Wv,
                                  __bf16* __restrict__ Wt) {
  int idx = blockIdx.x * blockDim.x + threadIdx.x;  // 3072*2048
  int n = idx >> 11;
  int k = idx & 2047;
  float v;
  if (n < 2048)      v = Wq[(size_t)k * 2048 + n];
  else if (n < 2560) v = Wk[(size_t)k * 512 + (n - 2048)];
  else               v = Wv[(size_t)k * 512 + (n - 2560)];
  Wt[(size_t)n * 2048 + k] = (__bf16)v;
}

__global__ void build_bqkv_kernel(const float* __restrict__ bq,
                                  const float* __restrict__ bk,
                                  const float* __restrict__ bv,
                                  float* __restrict__ bqkv) {
  int n = blockIdx.x * blockDim.x + threadIdx.x;
  if (n >= 3072) return;
  bqkv[n] = (n < 2048) ? bq[n] : (n < 2560 ? bk[n - 2048] : bv[n - 2560]);
}

__global__ void build_wo_kernel(const float* __restrict__ Wo,
                                __bf16* __restrict__ Wt) {
  int idx = blockIdx.x * blockDim.x + threadIdx.x;  // 2048*2048
  int n = idx >> 11;
  int k = idx & 2047;
  Wt[(size_t)n * 2048 + k] = (__bf16)Wo[(size_t)k * 2048 + n];
}

// ---------------------------------------------------------------------------
// Tiled bf16 WMMA GEMM with TDM tile staging.
// C[M,N] = A[M,K] @ B[K,N] + bias, B given transposed as Bt[N,K].
// Block tile 128x128, 8 waves (2x4), wave tile 64x32, K step 32.
// ---------------------------------------------------------------------------
template <bool STORE_F32>
__global__ void __launch_bounds__(256)
gemm_bf16_kernel(const __bf16* __restrict__ A, int lda,
                 const __bf16* __restrict__ Bt, int ldb,
                 const float* __restrict__ bias,
                 float* __restrict__ outF, __bf16* __restrict__ outB, int ldo,
                 int M, int N, int K) {
  __shared__ __bf16 sA[128 * 32];
  __shared__ __bf16 sB[128 * 32];

  const int tid  = threadIdx.x;
  const int wave = tid >> 5;
  const int lane = tid & 31;
  const int l15  = lane & 15;
  const int half = lane >> 4;
  const int wm   = wave & 1;   // M offset wm*64
  const int wn   = wave >> 1;  // N offset wn*32
  const int m0   = blockIdx.y * 128;
  const int n0   = blockIdx.x * 128;

  const uint32_t sA_off = (uint32_t)(uintptr_t)&sA[0];  // LDS byte offset
  const uint32_t sB_off = (uint32_t)(uintptr_t)&sB[0];

  v8f acc[4][2];
#pragma unroll
  for (int ti = 0; ti < 4; ++ti)
#pragma unroll
    for (int tj = 0; tj < 2; ++tj) acc[ti][tj] = zero8();

  for (int k0 = 0; k0 < K; k0 += 32) {
    __syncthreads();
    if (wave == 0) {
      // TDM: A tile [128 rows][32 cols], Bt tile [128 rows][32 cols].
      tdm_load_tile_2d_bf16(A  + (size_t)m0 * lda + k0, sA_off,
                            32, 128, (uint32_t)K, (uint32_t)M, (uint32_t)lda);
      tdm_load_tile_2d_bf16(Bt + (size_t)n0 * ldb + k0, sB_off,
                            32, 128, (uint32_t)K, (uint32_t)N, (uint32_t)ldb);
      wait_tensorcnt0();
    }
    __syncthreads();

    if (k0 + 32 < K) {  // gfx1250 global_prefetch_b8 for the next K tile
      __builtin_prefetch(A  + (size_t)(m0 + (tid >> 1)) * lda + k0 + 32, 0, 1);
      __builtin_prefetch(Bt + (size_t)(n0 + (tid >> 1)) * ldb + k0 + 32, 0, 1);
    }

    // Hoist all fragments, then issue the 8 WMMAs back-to-back.
    v16bf bfr[2];
#pragma unroll
    for (int tj = 0; tj < 2; ++tj) {
      const __bf16* bp = sB + (wn * 32 + tj * 16 + l15) * 32;
      bfr[tj] = join16(*(const v8bf*)(bp + half * 8), *(const v8bf*)(bp + 16 + half * 8));
    }
    v16bf afr[4];
#pragma unroll
    for (int ti = 0; ti < 4; ++ti) {
      const __bf16* ap = sA + (wm * 64 + ti * 16 + l15) * 32;
      afr[ti] = join16(*(const v8bf*)(ap + half * 8),
                       *(const v8bf*)(ap + 16 + half * 8));
    }
#pragma unroll
    for (int ti = 0; ti < 4; ++ti) {
      acc[ti][0] = WMMA_BF16(afr[ti], bfr[0], acc[ti][0]);
      acc[ti][1] = WMMA_BF16(afr[ti], bfr[1], acc[ti][1]);
    }
  }

  // Epilogue: c[r] -> row (r + 8*half), col (lane&15) of the 16x16 tile.
#pragma unroll
  for (int ti = 0; ti < 4; ++ti)
#pragma unroll
    for (int tj = 0; tj < 2; ++tj) {
      int col = n0 + wn * 32 + tj * 16 + l15;
      float bv = bias[col];
#pragma unroll
      for (int r = 0; r < 8; ++r) {
        int row = m0 + wm * 64 + ti * 16 + r + 8 * half;
        float v = acc[ti][tj][r] + bv;
        if (STORE_F32) outF[(size_t)row * ldo + col] = v;
        else           outB[(size_t)row * ldo + col] = (__bf16)v;
      }
    }
}

// ---------------------------------------------------------------------------
// Flash attention (causal GQA).  Grid: (S/64, H, B).  Block: 128 (4 waves).
// ---------------------------------------------------------------------------
__global__ void __launch_bounds__(128)
gqa_attn_kernel(const __bf16* __restrict__ QKV,
                const float* __restrict__ pad,  // [B][S]
                __bf16* __restrict__ AO) {      // [B*S][2048]
  __shared__ __bf16 sK[32 * 128];      // [kv][dh]
  __shared__ __bf16 sVt[128 * 32];     // [dh][kv]
  __shared__ __bf16 sP[4][16 * 32];    // per-wave P scratch [q][kv]

  const int tid  = threadIdx.x;
  const int wave = tid >> 5;
  const int lane = tid & 31;
  const int l15  = lane & 15;
  const int half = lane >> 4;
  const int q0   = blockIdx.x * 64;
  const int h    = blockIdx.y;
  const int b    = blockIdx.z;
  const int g    = h >> 2;          // KV group (4 heads per group)
  const int qrw  = q0 + wave * 16;  // this wave's first query row
  const size_t ldq = 3072;

  const __bf16* Qbase = QKV + (size_t)b * 2048 * ldq + (size_t)h * 128;
  const __bf16* Kbase = QKV + (size_t)b * 2048 * ldq + 2048 + (size_t)g * 128;
  const __bf16* Vbase = Kbase + 512;

  // Wave's Q fragments: 4 A-frags covering DH=128.
  v16bf qf[4];
#pragma unroll
  for (int kt = 0; kt < 4; ++kt) {
    const __bf16* qp = Qbase + (size_t)(qrw + l15) * ldq + kt * 32;
    qf[kt] = join16(*(const v8bf*)(qp + half * 8),
                    *(const v8bf*)(qp + 16 + half * 8));
  }

  v8f o[8];
#pragma unroll
  for (int f = 0; f < 8; ++f) o[f] = zero8();
  float m_prev[8], l_sum[8];
#pragma unroll
  for (int r = 0; r < 8; ++r) { m_prev[r] = -1.0e30f; l_sum[r] = 0.0f; }

  const float scale = 0.088388347648318447f;  // 1/sqrt(128)
  const int kv_end = q0 + 64;                 // causal bound for this block

  for (int kv0 = 0; kv0 < kv_end; kv0 += 32) {
    __syncthreads();
    // Stage K tile [32][128] (async to LDS when available) and V transposed.
#pragma unroll
    for (int uu = 0; uu < 4; ++uu) {
      int u = tid + uu * 128;  // 512 uint4 total
      int r = u >> 4;          // kv row
      int c = u & 15;          // uint4 within 128-col row
#ifdef HAS_ASYNC_LDS
      __builtin_amdgcn_global_load_async_to_lds_b128(
          (gptr_b128)(const void*)(((const uint4*)(Kbase + (size_t)(kv0 + r) * ldq)) + c),
          (lptr_b128)(void*)(((uint4*)sK) + u), 0, 0);
#else
      ((uint4*)sK)[u] = ((const uint4*)(Kbase + (size_t)(kv0 + r) * ldq))[c];
#endif
      const __bf16* vp = Vbase + (size_t)(kv0 + r) * ldq + c * 8;
#pragma unroll
      for (int e = 0; e < 8; ++e) sVt[(c * 8 + e) * 32 + r] = vp[e];
    }
#ifdef HAS_ASYNC_LDS
    asm volatile("s_wait_asynccnt 0x0" ::: "memory");
#endif
    __syncthreads();

    // S = Q @ K^T : hoist all 8 B-frags, then 8 WMMAs.
    v16bf kb[8];
#pragma unroll
    for (int kt = 0; kt < 4; ++kt) {
      const __bf16* kp0 = sK + l15 * 128 + kt * 32 + half * 8;
      const __bf16* kp1 = sK + (16 + l15) * 128 + kt * 32 + half * 8;
      kb[2 * kt]     = join16(((const v8bf*)kp0)[0], ((const v8bf*)kp0)[2]);
      kb[2 * kt + 1] = join16(((const v8bf*)kp1)[0], ((const v8bf*)kp1)[2]);
    }
    v8f s0 = zero8(), s1 = zero8();
#pragma unroll
    for (int kt = 0; kt < 4; ++kt) {
      s0 = WMMA_BF16(qf[kt], kb[2 * kt],     s0);
      s1 = WMMA_BF16(qf[kt], kb[2 * kt + 1], s1);
    }

    // Scale + causal + padding masks, online softmax update.
    float pad0 = pad[(size_t)b * 2048 + kv0 + l15] * -1.0e9f;
    float pad1 = pad[(size_t)b * 2048 + kv0 + 16 + l15] * -1.0e9f;
    int c0 = kv0 + l15, c1 = kv0 + 16 + l15;
#pragma unroll
    for (int r = 0; r < 8; ++r) {
      int q = qrw + r + 8 * half;
      float v0 = s0[r] * scale + pad0 + (c0 > q ? -1.0e9f : 0.0f);
      float v1 = s1[r] * scale + pad1 + (c1 > q ? -1.0e9f : 0.0f);
      float mx = fmaxf(v0, v1);
      mx = fmaxf(mx, __shfl_xor(mx, 1, 32));
      mx = fmaxf(mx, __shfl_xor(mx, 2, 32));
      mx = fmaxf(mx, __shfl_xor(mx, 4, 32));
      mx = fmaxf(mx, __shfl_xor(mx, 8, 32));
      float mn = fmaxf(m_prev[r], mx);
      float alpha = __expf(m_prev[r] - mn);
      m_prev[r] = mn;
      float p0 = __expf(v0 - mn);
      float p1 = __expf(v1 - mn);
      float rs = p0 + p1;
      rs += __shfl_xor(rs, 1, 32);
      rs += __shfl_xor(rs, 2, 32);
      rs += __shfl_xor(rs, 4, 32);
      rs += __shfl_xor(rs, 8, 32);
      l_sum[r] = l_sum[r] * alpha + rs;
#pragma unroll
      for (int f = 0; f < 8; ++f) o[f][r] *= alpha;
      sP[wave][(r + 8 * half) * 32 + l15]      = (__bf16)p0;
      sP[wave][(r + 8 * half) * 32 + 16 + l15] = (__bf16)p1;
    }
    __syncthreads();

    // Reload P as A-fragment, hoist all V B-frags, then 8 WMMAs.
    const __bf16* pp = &sP[wave][l15 * 32];
    v16bf pa = join16(*(const v8bf*)(pp + half * 8),
                      *(const v8bf*)(pp + 16 + half * 8));
    v16bf vb[8];
#pragma unroll
    for (int f = 0; f < 8; ++f) {
      const __bf16* vp = sVt + (f * 16 + l15) * 32 + half * 8;
      vb[f] = join16(((const v8bf*)vp)[0], ((const v8bf*)vp)[2]);
    }
#pragma unroll
    for (int f = 0; f < 8; ++f) o[f] = WMMA_BF16(pa, vb[f], o[f]);
  }

  // Normalize and store attention output (bf16 for the final GEMM).
#pragma unroll
  for (int r = 0; r < 8; ++r) {
    float inv = 1.0f / l_sum[r];
    size_t row = (size_t)b * 2048 + qrw + r + 8 * half;
#pragma unroll
    for (int f = 0; f < 8; ++f)
      AO[row * 2048 + h * 128 + f * 16 + l15] = (__bf16)(o[f][r] * inv);
  }
}

// ---------------------------------------------------------------------------
// Host-side orchestration
// ---------------------------------------------------------------------------
static constexpr size_t OFF_XBF  = 0;                              // 4096x2048 bf16
static constexpr size_t OFF_WQKV = OFF_XBF  + 4096ull * 2048 * 2;  // 3072x2048 bf16
static constexpr size_t OFF_BQKV = OFF_WQKV + 3072ull * 2048 * 2;  // 3072 f32
static constexpr size_t OFF_WO   = OFF_BQKV + 3072ull * 4;         // 2048x2048 bf16
static constexpr size_t OFF_QKV  = OFF_WO   + 2048ull * 2048 * 2;  // 4096x3072 bf16
static constexpr size_t OFF_AO   = OFF_QKV  + 4096ull * 3072 * 2;  // 4096x2048 bf16

extern "C" void kernel_launch(void* const* d_in, const int* in_sizes, int n_in,
                              void* d_out, int out_size, void* d_ws, size_t ws_size,
                              hipStream_t stream) {
  const float* X   = (const float*)d_in[0];
  // d_in[1] = causal mask (computed analytically on device; not read)
  const float* pad = (const float*)d_in[2];
  const float* Wq  = (const float*)d_in[3];
  const float* bq  = (const float*)d_in[4];
  const float* Wk  = (const float*)d_in[5];
  const float* bk  = (const float*)d_in[6];
  const float* Wv  = (const float*)d_in[7];
  const float* bv  = (const float*)d_in[8];
  const float* Wo  = (const float*)d_in[9];
  const float* bo  = (const float*)d_in[10];

  char* ws = (char*)d_ws;
  __bf16* Xbf   = (__bf16*)(ws + OFF_XBF);
  __bf16* Wqkvt = (__bf16*)(ws + OFF_WQKV);
  float*  bqkv  = (float*) (ws + OFF_BQKV);
  __bf16* Wot   = (__bf16*)(ws + OFF_WO);
  __bf16* QKV   = (__bf16*)(ws + OFF_QKV);
  __bf16* AO    = (__bf16*)(ws + OFF_AO);

  // 1) fp32 -> bf16 conversions / weight packing
  f32_to_bf16_kernel<<<(4096 * 2048) / 256, 256, 0, stream>>>(X, Xbf, 4096 * 2048);
  build_wqkv_kernel<<<(3072 * 2048) / 256, 256, 0, stream>>>(Wq, Wk, Wv, Wqkvt);
  build_bqkv_kernel<<<12, 256, 0, stream>>>(bq, bk, bv, bqkv);
  build_wo_kernel<<<(2048 * 2048) / 256, 256, 0, stream>>>(Wo, Wot);

  // 2) QKV = X @ [Wq|Wk|Wv] + b   -> bf16 [4096][3072]
  gemm_bf16_kernel<false><<<dim3(3072 / 128, 4096 / 128), 256, 0, stream>>>(
      Xbf, 2048, Wqkvt, 2048, bqkv, nullptr, QKV, 3072, 4096, 3072, 2048);

  // 3) causal GQA flash attention -> AO bf16 [4096][2048]
  gqa_attn_kernel<<<dim3(2048 / 64, 16, 2), 128, 0, stream>>>(QKV, pad, AO);

  // 4) out = AO @ Wo + bo -> f32 d_out
  gemm_bf16_kernel<true><<<dim3(2048 / 128, 4096 / 128), 256, 0, stream>>>(
      AO, 2048, Wot, 2048, bo, (float*)d_out, nullptr, 2048, 4096, 2048, 2048);
}
